// MixtureOfExperts_11785390260755
// MI455X (gfx1250) — matrix operand order repaired and checked by reading, not compiled, not run on hardware
//
#include <hip/hip_runtime.h>
#include <hip/hip_bf16.h>

typedef __attribute__((ext_vector_type(16))) __bf16 v16bf;
typedef __attribute__((ext_vector_type(8)))  float  v8f;
typedef __attribute__((ext_vector_type(4)))  int    v4i_t;

#define NTOK   16384
#define CDIM   768
#define IDIM   1536
#define NEXP   16
#define MT     32          // tokens per block tile

#if defined(__has_builtin)
#if __has_builtin(__builtin_amdgcn_global_load_async_to_lds_b128) && \
    __has_builtin(__builtin_amdgcn_s_wait_asynccnt)
#define USE_ASYNC_GATHER 1
#endif
#endif

__device__ __forceinline__ unsigned short f32_bf16(float f) {
    unsigned int u = __builtin_bit_cast(unsigned int, f);
    unsigned int r = (u + 0x7FFFu + ((u >> 16) & 1u)) >> 16;   // RNE
    return (unsigned short)r;
}

union BV { uint4 q[2]; v16bf v; };

// ---------------- weight transpose + f32->bf16 convert -------------------
// src: [batch][R][Cc] f32 row-major ; dst: [batch][Cc][R] bf16 row-major
__global__ __launch_bounds__(256) void transpose_cvt_kernel(
    const float* __restrict__ src, unsigned short* __restrict__ dst,
    int R, int Cc)
{
    __shared__ float tile[64][65];
    const int b  = blockIdx.z;
    const float*         s = src + (size_t)b * R * Cc;
    unsigned short*      d = dst + (size_t)b * R * Cc;
    const int c0 = blockIdx.x * 64, r0 = blockIdx.y * 64;
    const int tx = threadIdx.x & 63, ty = threadIdx.x >> 6;   // 64 x 4
    #pragma unroll
    for (int i = ty; i < 64; i += 4)
        tile[i][tx] = s[(size_t)(r0 + i) * Cc + c0 + tx];
    __syncthreads();
    #pragma unroll
    for (int i = ty; i < 64; i += 4)
        d[(size_t)(c0 + i) * R + r0 + tx] = f32_bf16(tile[tx][i]);
}

// ---------------- x f32 -> bf16 ------------------------------------------
__global__ __launch_bounds__(256) void cvt_x_kernel(
    const float* __restrict__ x, unsigned short* __restrict__ xb)
{
    const int i = blockIdx.x * 256 + threadIdx.x;
    float4 v = ((const float4*)x)[i];
    ushort4 o;
    o.x = f32_bf16(v.x); o.y = f32_bf16(v.y);
    o.z = f32_bf16(v.z); o.w = f32_bf16(v.w);
    ((ushort4*)xb)[i] = o;
}

// ---------------- router: logits, softmax top-2, counting sort -----------
__global__ __launch_bounds__(256) void router_kernel(
    const float* __restrict__ x, const float* __restrict__ wgate,
    const float* __restrict__ bias,
    int* __restrict__ cnt, int* __restrict__ tokL, float* __restrict__ wtL)
{
    __shared__ float lg[16][16];
    const int tile = blockIdx.x;
    const int tl = threadIdx.x >> 4;       // token-in-tile
    const int e  = threadIdx.x & 15;       // expert
    {
        const float* xr = x + (size_t)(tile * 16 + tl) * CDIM;
        float acc = 0.f;
        for (int c = 0; c < CDIM; ++c)
            acc += xr[c] * wgate[c * NEXP + e];
        lg[tl][e] = acc + bias[e];
    }
    __syncthreads();
    if (threadIdx.x < 16) {
        const int t = threadIdx.x;
        const int token = tile * 16 + t;
        float best = -1e30f; int bi = 0;
        #pragma unroll
        for (int k = 0; k < NEXP; ++k) {
            float v = lg[t][k];
            if (v > best) { best = v; bi = k; }
        }
        float b2v = -1e30f; int b2 = 0;
        #pragma unroll
        for (int k = 0; k < NEXP; ++k) {
            if (k == bi) continue;
            float v = lg[t][k];
            if (v > b2v) { b2v = v; b2 = k; }
        }
        const float r  = __expf(b2v - best);
        const float w0 = 1.f / (1.f + r);
        const float w1 = r   / (1.f + r);
        int s0 = atomicAdd(&cnt[bi], 1);
        tokL[(size_t)bi * NTOK + s0] = token; wtL[(size_t)bi * NTOK + s0] = w0;
        int s1 = atomicAdd(&cnt[b2], 1);
        tokL[(size_t)b2 * NTOK + s1] = token; wtL[(size_t)b2 * NTOK + s1] = w1;
    }
}

// ---------------- fused SwiGLU expert FFN (bf16 WMMA, M=32) --------------
// One block = one 32-token tile of one expert. 8 waves.
// Each B (weight) fragment is loaded once and reused for 2 M-sub-tiles.
// wgT/wuT : [e][I][C] bf16 ; wdT : [e][C][I] bf16
template<bool ROUTED>
__global__ __launch_bounds__(256, 1) void ffn_kernel(
    const unsigned short* __restrict__ xb,
    const unsigned short* __restrict__ wgT,
    const unsigned short* __restrict__ wuT,
    const unsigned short* __restrict__ wdT,
    const int* __restrict__ cnt, const int* __restrict__ tokL,
    const float* __restrict__ wtL,
    float* __restrict__ out)
{
    constexpr int SAS = CDIM + 8;          // 776  (16B-aligned row stride)
    constexpr int CH  = 256;               // I-chunk
    constexpr int SHS = CH + 8;            // 264
    __shared__ unsigned short sA[MT * SAS];
    __shared__ unsigned short sH[MT * SHS];
    __shared__ int   sTok[MT];
    __shared__ float sW[MT];

    const int e    = ROUTED ? blockIdx.y : 0;
    const int tile = blockIdx.x;
    const int count = ROUTED ? cnt[e] : NTOK;
    if (tile * MT >= count) return;

    const unsigned short* wg_e = wgT + (size_t)e * IDIM * CDIM;
    const unsigned short* wu_e = wuT + (size_t)e * IDIM * CDIM;
    const unsigned short* wd_e = wdT + (size_t)e * CDIM * IDIM;

    const int tid = threadIdx.x;
    if (tid < MT) {
        const int slot = tile * MT + tid;
        int token; float w;
        if (ROUTED) {
            if (slot < count) {
                token = tokL[(size_t)e * NTOK + slot];
                w     = wtL[(size_t)e * NTOK + slot];
            } else { token = 0; w = 0.f; }
        } else { token = slot; w = 1.f; }
        sTok[tid] = token; sW[tid] = w;
    }
    __syncthreads();

    // gather MT x 768 bf16 x-tile into LDS (16B chunks)
#if USE_ASYNC_GATHER
    for (int cc = tid; cc < MT * (CDIM / 8); cc += 256) {
        const int row = cc / (CDIM / 8);
        const int off = (cc % (CDIM / 8)) * 8;
        const unsigned short* gp = xb + (size_t)sTok[row] * CDIM + off;
        __builtin_amdgcn_global_load_async_to_lds_b128(
            (__attribute__((address_space(1))) v4i_t*)gp,
            (__attribute__((address_space(3))) v4i_t*)(sA + row * SAS + off),
            0, 0);
    }
    __builtin_amdgcn_s_wait_asynccnt(0);
#else
    for (int cc = tid; cc < MT * (CDIM / 8); cc += 256) {
        const int row = cc / (CDIM / 8);
        const int off = (cc % (CDIM / 8)) * 8;
        uint4 v = *(const uint4*)(xb + (size_t)sTok[row] * CDIM + off);
        *(uint4*)(sA + row * SAS + off) = v;
    }
#endif
    __syncthreads();

    const int lane = tid & 31, wave = tid >> 5;
    const int m   = lane & 15;             // A row (in subtile) / B col / D col
    const int sel = lane >> 4;             // lane-half selector

    v8f acc_y[2][6];
    #pragma unroll
    for (int s = 0; s < 2; ++s)
        #pragma unroll
        for (int i = 0; i < 6; ++i) acc_y[s][i] = (v8f){0,0,0,0,0,0,0,0};

    for (int chunk = 0; chunk < IDIM / CH; ++chunk) {
        if (chunk) __syncthreads();
        const int cb = chunk * CH;

        // ---- GEMM1: g = x@wg, u = x@wu  (2 col-tiles/wave, 2 M-subtiles) ----
        #pragma unroll
        for (int t = 0; t < 2; ++t) {
            const int ic0 = (wave * 2 + t) * 16;       // col in chunk
            const int i0  = cb + ic0;                  // global I col
            v8f ag0 = (v8f){0,0,0,0,0,0,0,0};
            v8f au0 = (v8f){0,0,0,0,0,0,0,0};
            v8f ag1 = (v8f){0,0,0,0,0,0,0,0};
            v8f au1 = (v8f){0,0,0,0,0,0,0,0};
            const unsigned short* wgr = wg_e + (size_t)(i0 + m) * CDIM;
            const unsigned short* wur = wu_e + (size_t)(i0 + m) * CDIM;
            for (int ks = 0; ks < CDIM / 32; ++ks) {
                const int k0 = ks * 32;
                BV a0, a1;
                a0.q[0] = *(const uint4*)(sA + m * SAS + k0 + sel * 16);
                a0.q[1] = *(const uint4*)(sA + m * SAS + k0 + sel * 16 + 8);
                a1.q[0] = *(const uint4*)(sA + (16 + m) * SAS + k0 + sel * 16);
                a1.q[1] = *(const uint4*)(sA + (16 + m) * SAS + k0 + sel * 16 + 8);
                BV bg;
                const uint4* pg = (const uint4*)(wgr + k0 + sel * 16);
                bg.q[0] = pg[0]; bg.q[1] = pg[1];
                ag0 = __builtin_amdgcn_wmma_f32_16x16x32_bf16(
                          false, a0.v, false, bg.v, (short)0, ag0, false, false);
                ag1 = __builtin_amdgcn_wmma_f32_16x16x32_bf16(
                          false, a1.v, false, bg.v, (short)0, ag1, false, false);
                BV bu;
                const uint4* pu = (const uint4*)(wur + k0 + sel * 16);
                bu.q[0] = pu[0]; bu.q[1] = pu[1];
                au0 = __builtin_amdgcn_wmma_f32_16x16x32_bf16(
                          false, a0.v, false, bu.v, (short)0, au0, false, false);
                au1 = __builtin_amdgcn_wmma_f32_16x16x32_bf16(
                          false, a1.v, false, bu.v, (short)0, au1, false, false);
            }
            // h = silu(g) * u ; store bf16 tiles to LDS (D layout -> row-major)
            #pragma unroll
            for (int r = 0; r < 8; ++r) {
                float g = ag0[r], u = au0[r];
                sH[(r + 8 * sel) * SHS + ic0 + m] =
                    f32_bf16((g / (1.f + __expf(-g))) * u);
                g = ag1[r]; u = au1[r];
                sH[(16 + r + 8 * sel) * SHS + ic0 + m] =
                    f32_bf16((g / (1.f + __expf(-g))) * u);
            }
        }
        __syncthreads();

        // ---- GEMM2: y += h @ wd  (6 n-tiles/wave, B reused for 2 subtiles) ----
        #pragma unroll
        for (int nt = 0; nt < 6; ++nt) {
            const int n0 = (wave * 6 + nt) * 16;       // global out col base
            const unsigned short* wdr = wd_e + (size_t)(n0 + m) * IDIM + cb;
            v8f acc0 = acc_y[0][nt];
            v8f acc1 = acc_y[1][nt];
            for (int ks = 0; ks < CH / 32; ++ks) {
                const int k0 = ks * 32;
                BV b;
                const uint4* pb = (const uint4*)(wdr + k0 + sel * 16);
                b.q[0] = pb[0]; b.q[1] = pb[1];
                BV a0, a1;
                a0.q[0] = *(const uint4*)(sH + m * SHS + k0 + sel * 16);
                a0.q[1] = *(const uint4*)(sH + m * SHS + k0 + sel * 16 + 8);
                a1.q[0] = *(const uint4*)(sH + (16 + m) * SHS + k0 + sel * 16);
                a1.q[1] = *(const uint4*)(sH + (16 + m) * SHS + k0 + sel * 16 + 8);
                acc0 = __builtin_amdgcn_wmma_f32_16x16x32_bf16(
                           false, a0.v, false, b.v, (short)0, acc0, false, false);
                acc1 = __builtin_amdgcn_wmma_f32_16x16x32_bf16(
                           false, a1.v, false, b.v, (short)0, acc1, false, false);
            }
            acc_y[0][nt] = acc0;
            acc_y[1][nt] = acc1;
        }
    }

    // ---- write / accumulate output ----
    #pragma unroll
    for (int s = 0; s < 2; ++s) {
        #pragma unroll
        for (int nt = 0; nt < 6; ++nt) {
            const int n = (wave * 6 + nt) * 16 + m;
            #pragma unroll
            for (int r = 0; r < 8; ++r) {
                const int row = s * 16 + r + 8 * sel;
                const float w = sW[row];
                const float val = w * acc_y[s][nt][r];
                const size_t o = (size_t)sTok[row] * CDIM + n;
                if (ROUTED) {
                    if (w != 0.f) atomicAdd(&out[o], val);
                } else {
                    out[o] = val;
                }
            }
        }
    }
}

// -------------------------------------------------------------------------
extern "C" void kernel_launch(void* const* d_in, const int* in_sizes, int n_in,
                              void* d_out, int out_size, void* d_ws, size_t ws_size,
                              hipStream_t stream)
{
    const float* x      = (const float*)d_in[0];
    const float* w_gate = (const float*)d_in[1];
    const float* e_bias = (const float*)d_in[2];
    const float* wg     = (const float*)d_in[3];
    const float* wu     = (const float*)d_in[4];
    const float* wd     = (const float*)d_in[5];
    const float* swg    = (const float*)d_in[6];
    const float* swu    = (const float*)d_in[7];
    const float* swd    = (const float*)d_in[8];
    float* out = (float*)d_out;

    // workspace carve-out (all 256B aligned)
    char* ws = (char*)d_ws;
    auto carve = [&ws](size_t bytes) {
        void* p = (void*)ws;
        ws += (bytes + 255) & ~(size_t)255;
        return p;
    };
    unsigned short* xb   = (unsigned short*)carve((size_t)NTOK * CDIM * 2);
    unsigned short* wgT  = (unsigned short*)carve((size_t)NEXP * CDIM * IDIM * 2);
    unsigned short* wuT  = (unsigned short*)carve((size_t)NEXP * CDIM * IDIM * 2);
    unsigned short* wdT  = (unsigned short*)carve((size_t)NEXP * CDIM * IDIM * 2);
    unsigned short* swgT = (unsigned short*)carve((size_t)CDIM * IDIM * 2);
    unsigned short* swuT = (unsigned short*)carve((size_t)CDIM * IDIM * 2);
    unsigned short* swdT = (unsigned short*)carve((size_t)CDIM * IDIM * 2);
    int*   cnt  = (int*)  carve(NEXP * sizeof(int));
    int*   tokL = (int*)  carve((size_t)NEXP * NTOK * sizeof(int));
    float* wtL  = (float*)carve((size_t)NEXP * NTOK * sizeof(float));

    (void)hipMemsetAsync(cnt, 0, NEXP * sizeof(int), stream);

    // weight conversion (transposed for contiguous WMMA B-operand loads)
    transpose_cvt_kernel<<<dim3(IDIM/64, CDIM/64, NEXP), 256, 0, stream>>>(wg,  wgT,  CDIM, IDIM);
    transpose_cvt_kernel<<<dim3(IDIM/64, CDIM/64, NEXP), 256, 0, stream>>>(wu,  wuT,  CDIM, IDIM);
    transpose_cvt_kernel<<<dim3(CDIM/64, IDIM/64, NEXP), 256, 0, stream>>>(wd,  wdT,  IDIM, CDIM);
    transpose_cvt_kernel<<<dim3(IDIM/64, CDIM/64, 1),    256, 0, stream>>>(swg, swgT, CDIM, IDIM);
    transpose_cvt_kernel<<<dim3(IDIM/64, CDIM/64, 1),    256, 0, stream>>>(swu, swuT, CDIM, IDIM);
    transpose_cvt_kernel<<<dim3(CDIM/64, IDIM/64, 1),    256, 0, stream>>>(swd, swdT, IDIM, CDIM);

    cvt_x_kernel<<<(NTOK * CDIM / 4) / 256, 256, 0, stream>>>(x, xb);

    router_kernel<<<NTOK / 16, 256, 0, stream>>>(x, w_gate, e_bias, cnt, tokL, wtL);

    // shared expert: initializes out with plain stores
    ffn_kernel<false><<<dim3(NTOK / MT, 1), 256, 0, stream>>>(
        xb, swgT, swuT, swdT, nullptr, nullptr, nullptr, out);

    // routed experts: weighted atomic accumulation
    ffn_kernel<true><<<dim3(NTOK / MT, NEXP), 256, 0, stream>>>(
        xb, wgT, wuT, wdT, cnt, tokL, wtL, out);
}
